// LinearCDE_76398878261531
// MI455X (gfx1250) — matrix operand order repaired and checked by reading, not compiled, not run on hardware
//
#include <hip/hip_runtime.h>

typedef __attribute__((ext_vector_type(16))) __bf16 v16bf;
typedef __attribute__((ext_vector_type(8)))  float  v8f;

// Problem constants
#define BB   16
#define LL   2048
#define TT   2047      // scan steps
#define DD   63
#define HH   128
#define OO   64        // OM == D+1 == XI
#define RR   (HH*HH)   // 16384 rows of A-hat = (h,j)
#define CH   16        // chunk of time steps == one WMMA N tile
#define NCH  128       // 128 chunks * 16 = 2048 >= 2047
#define NTILE (RR/16)  // 1024 row tiles

// ---------------------------------------------------------------------------
// Prep 1: swizzle vf_A (H,OM,H) f32 -> bf16 A-fragments.
// Linear layout: idx = ((tile*2 + ks)*32 + lane)*16 + elem
// A-frag element mapping (16-bit A 16x32, wave32):
//   M = lane%16 ; half = lane/16
//   K = (elem<8) ? 8*half + elem : 16 + 8*half + (elem-8)   (within ks-th K32 slab)
// Row r = tile*16 + M = h*128 + j ; o = ks*32 + K ; src = vf_A[h][o][j]
// ---------------------------------------------------------------------------
__global__ void lcde_prep_A(const float* __restrict__ vfA, __bf16* __restrict__ Abf) {
    int idx  = blockIdx.x * 256 + threadIdx.x;          // 0 .. RR*OO-1 (1,048,576)
    int elem = idx & 15;
    int lane = (idx >> 4) & 31;
    int ks   = (idx >> 9) & 1;
    int tile = idx >> 10;
    int half = lane >> 4;
    int kk   = (elem < 8) ? (8 * half + elem) : (16 + 8 * half + (elem - 8));
    int o    = ks * 32 + kk;
    int r    = tile * 16 + (lane & 15);
    int h    = r >> 7;
    int j    = r & 127;
    Abf[idx] = (__bf16)vfA[(h * OO + o) * HH + j];
}

// ---------------------------------------------------------------------------
// Prep 2: bterm[b][t][h] = sum_x w_xi[b,t,x] * vf_B[h,x],
// w_xi[b,t,0]=ts[t+1]=(t+1)/2047, w_xi[b,t,x>=1]=X[b,t+1,x-1]
// ---------------------------------------------------------------------------
__global__ void lcde_prep_bterm(const float* __restrict__ X,
                                const float* __restrict__ vfB,
                                float* __restrict__ bt) {
    int idx = blockIdx.x * 256 + threadIdx.x;           // 0 .. BB*TT*HH-1
    if (idx >= BB * TT * HH) return;
    int h = idx & 127;
    int t = (idx >> 7) % TT;
    int b = idx / (HH * TT);
    const float* xr = X + ((size_t)b * LL + (t + 1)) * DD;
    const float* vb = vfB + h * OO;
    float s = vb[0] * ((float)(t + 1) * (1.0f / 2047.0f));
    for (int x = 1; x < OO; ++x) s += vb[x] * xr[x - 1];
    bt[idx] = s;
}

// ---------------------------------------------------------------------------
// Main: one workgroup per batch. Chunked: WMMA GEMM to form M_t for 16 steps,
// then 16 sequential LDS matvec steps of the linear recurrence.
// ---------------------------------------------------------------------------
__global__ void __launch_bounds__(1024)
lcde_main(const float* __restrict__ X, const float* __restrict__ initM,
          const float* __restrict__ initB, const float* __restrict__ bterm,
          const __bf16* __restrict__ Abf, float* __restrict__ Mws,
          float* __restrict__ out) {
    __shared__ float y[HH];
    __shared__ float part[1024];

    const int b    = blockIdx.x;
    const int tid  = threadIdx.x;
    const int lane = tid & 31;
    const int wid  = tid >> 5;
    const int half = lane >> 4;
    const int col  = lane & 15;          // N index = step-in-chunk

    const float* Xb = X + (size_t)b * LL * DD;
    float* M = Mws + (size_t)b * (CH * RR);   // [s][r] layout, 1 MiB per batch

    // y0 = X[b,0,:] @ initM^T + bias
    if (tid < HH) {
        float s = initB[tid];
        const float* wrow = initM + tid * DD;
        for (int d = 0; d < DD; ++d) s += Xb[d] * wrow[d];
        y[tid] = s;
    }
    __syncthreads();

    const float invL = 1.0f / (float)LL;
    const v16bf* Afrag = (const v16bf*)Abf;

    #pragma unroll 1
    for (int c = 0; c < NCH; ++c) {
        // ---- Build B fragments (W chunk: o x t, 64x16 bf16) per wave ----
        // B 32x16 layout: N = lane%16, K = (i<8) ? 8*half + i : 16 + 8*half + (i-8)  (per 32-K slab ks)
        v16bf bf0, bf1;
        const int tg    = c * CH + col;               // global step for this column
        const bool vld  = (tg < TT);
        const float* xrow = Xb + (size_t)(tg + 1) * DD;
        const float tsv = (float)(tg + 1) * (1.0f / 2047.0f);
        #pragma unroll
        for (int i = 0; i < 16; ++i) {
            int o0 = (i < 8) ? (8 * half + i) : (16 + 8 * half + (i - 8));   // ks = 0
            int o1 = 32 + o0;                         // ks = 1
            float v0 = 0.0f, v1 = 0.0f;
            if (vld) {
                v0 = (o0 == 0) ? tsv : xrow[o0 - 1];
                v1 = xrow[o1 - 1];
            }
            bf0[i] = (__bf16)v0;
            bf1[i] = (__bf16)v1;
        }

        // ---- GEMM: Mstack[:, c-chunk] ; 32 row tiles per wave ----
        // unroll 1: keep per-iteration live set small (a0,a1,acc) so nothing
        // spills to scratch under the 1024-thread occupancy bound; 32 waves
        // per WGP hide the L2 fragment-load latency.
        #pragma unroll 1
        for (int mi = 0; mi < 32; ++mi) {
            int tile = wid * 32 + mi;
            const v16bf* ap = Afrag + (size_t)tile * 64 + lane;  // 2 frags * 32 lanes
            v16bf a0 = ap[0];
            v16bf a1 = ap[32];
            v8f acc = {};
            acc = __builtin_amdgcn_wmma_f32_16x16x32_bf16(false, a0, false, bf0,
                                                          (short)0, acc, false, false);
            acc = __builtin_amdgcn_wmma_f32_16x16x32_bf16(false, a1, false, bf1,
                                                          (short)0, acc, false, false);
            // D layout: row-in-tile = reg + 8*half, col = lane%16
            float* mout = M + (size_t)col * RR + tile * 16 + 8 * half;
            #pragma unroll
            for (int rg = 0; rg < 8; ++rg) mout[rg] = acc[rg];
        }
        __threadfence();      // make chunk's M visible to all waves (via L2)
        __syncthreads();

        // ---- Scan: y += (M_t y + bt)/L, sequential over steps in chunk ----
        const int nst = (TT - c * CH < CH) ? (TT - c * CH) : CH;
        const int h   = tid >> 3;
        const int seg = tid & 7;
        #pragma unroll 1
        for (int s = 0; s < nst; ++s) {
            const float* mr = M + (size_t)s * RR + h * HH + seg * 16;
            const float* yv = y + seg * 16;
            float p = 0.0f;
            #pragma unroll
            for (int j = 0; j < 16; ++j) p += mr[j] * yv[j];
            part[tid] = p;
            __syncthreads();
            if (tid < HH) {
                float tot = 0.0f;
                #pragma unroll
                for (int q = 0; q < 8; ++q) tot += part[tid * 8 + q];
                int t = c * CH + s;
                y[tid] += (tot + bterm[((size_t)b * TT + t) * HH + tid]) * invL;
            }
            __syncthreads();
        }
    }

    if (tid < HH) out[b * HH + tid] = y[tid];
}

// ---------------------------------------------------------------------------
extern "C" void kernel_launch(void* const* d_in, const int* in_sizes, int n_in,
                              void* d_out, int out_size, void* d_ws, size_t ws_size,
                              hipStream_t stream) {
    const float* X     = (const float*)d_in[0];   // (16,2048,63)
    const float* initM = (const float*)d_in[1];   // (128,63)
    const float* initB = (const float*)d_in[2];   // (128,)
    const float* vfA   = (const float*)d_in[3];   // (128,64,128)
    const float* vfB   = (const float*)d_in[4];   // (128,64)

    char* ws = (char*)d_ws;
    const size_t A_BYTES  = (size_t)RR * OO * sizeof(__bf16);        //  2,097,152
    const size_t BT_BYTES = (size_t)BB * TT * HH * sizeof(float);    // 16,769,024
    __bf16* Abf   = (__bf16*)ws;
    float*  bterm = (float*)(ws + A_BYTES);
    float*  Mws   = (float*)(ws + A_BYTES + BT_BYTES);               // 16 MiB

    lcde_prep_A<<<(RR * OO) / 256, 256, 0, stream>>>(vfA, Abf);
    lcde_prep_bterm<<<(BB * TT * HH + 255) / 256, 256, 0, stream>>>(X, vfB, bterm);
    lcde_main<<<BB, 1024, 0, stream>>>(X, initM, initB, bterm, Abf, Mws, (float*)d_out);
}
